// psRNN_14714557956638
// MI455X (gfx1250) — matrix operand order repaired and checked by reading, not compiled, not run on hardware
//
#include <hip/hip_runtime.h>

typedef __bf16 bf16_t;
typedef __attribute__((ext_vector_type(16))) __bf16 v16bf;
typedef __attribute__((ext_vector_type(8)))  float  v8f;

#define T_STEPS 512
#define BATCH   64
#define N_INP   64
#define N_HID   2048
#define N_OUT   64
#define NBLK    32

#define KB_HID  (N_HID / 32)   /* 64 K-blocks in recurrent GEMM */
#define KB_INP  (N_INP / 32)   /* 2 K-blocks in input projection */
#define FRAG_ELEMS 512         /* one 16x32 (or 32x16) bf16 fragment: 32 lanes x 16 elems */

#define WMMA_BF16(A_, B_, C_) \
  __builtin_amdgcn_wmma_f32_16x16x32_bf16(false, (A_), false, (B_), (short)0, (C_), false, false)

// ---------- fragment swizzle maps (element index inside a swizzled buffer) ----------
// B (K x N tile): lane n<16 holds K offsets {0..7,16..23}; lane n+16 holds {8..15,24..31}
static __device__ __forceinline__ size_t bswz(int h, int k, int KB) {
  const int ntile = h >> 4, n = h & 15;
  const int kb = k >> 5, kk = k & 31;
  const int lane = n + (((kk >> 3) & 1) << 4);
  const int e    = (kk & 7) + ((kk >> 4) << 3);
  return ((size_t)(ntile * KB + kb) * 32 + lane) * 16 + e;
}
// A (M x K tile): lane l<16 holds K offsets {0..7,16..23}; l>=16 holds {8..15,24..31}
static __device__ __forceinline__ size_t aswz(int row, int k, int KB) {
  const int m = row >> 4, r = row & 15;
  const int kb = k >> 5, kk = k & 31;
  const int lane = r + (((kk >> 3) & 1) << 4);
  const int e    = (kk & 7) + ((kk >> 4) << 3);
  return ((size_t)(m * KB + kb) * 32 + lane) * 16 + e;
}

// ---------- device-wide barrier across NBLK co-resident blocks ----------
static __device__ __forceinline__ void grid_barrier(unsigned* cnt, unsigned* gen) {
  __threadfence();
  __syncthreads();
  if (threadIdx.x == 0) {
    unsigned g   = __hip_atomic_load(gen, __ATOMIC_RELAXED, __HIP_MEMORY_SCOPE_AGENT);
    unsigned old = __hip_atomic_fetch_add(cnt, 1u, __ATOMIC_ACQ_REL, __HIP_MEMORY_SCOPE_AGENT);
    if (old == (unsigned)(NBLK - 1)) {
      __hip_atomic_store(cnt, 0u, __ATOMIC_RELAXED, __HIP_MEMORY_SCOPE_AGENT);
      __hip_atomic_fetch_add(gen, 1u, __ATOMIC_RELEASE, __HIP_MEMORY_SCOPE_AGENT);
    } else {
      while (__hip_atomic_load(gen, __ATOMIC_ACQUIRE, __HIP_MEMORY_SCOPE_AGENT) == g) {
        __builtin_amdgcn_s_sleep(2);
      }
    }
  }
  __syncthreads();
  __threadfence();
}

// ---------- prep: fp32 -> bf16 conversion into fragment-swizzled layouts ----------
__global__ void psrnn_prep(const float* __restrict__ Wh, const float* __restrict__ Wi,
                           const float* __restrict__ Wr, const float* __restrict__ x,
                           bf16_t* __restrict__ whb, bf16_t* __restrict__ wib,
                           bf16_t* __restrict__ wrb, bf16_t* __restrict__ xb,
                           float* __restrict__ sf0, bf16_t* __restrict__ sb0,
                           unsigned* __restrict__ bar) {
  const int NWH = N_HID * N_HID;          // 4194304 (largest)
  const int NWI = N_HID * N_INP;
  const int NWR = N_OUT * N_HID;
  const int NX  = T_STEPS * BATCH * N_INP;
  const int NS  = BATCH * N_HID;
  for (int i = blockIdx.x * blockDim.x + threadIdx.x; i < NWH;
       i += gridDim.x * blockDim.x) {
    { const int h = i >> 11, k = i & 2047; whb[bswz(h, k, KB_HID)] = (bf16_t)Wh[i]; }
    if (i < NX) {
      const int t = i >> 12, rem = i & 4095, row = rem >> 6, k = rem & 63;
      xb[(size_t)t * (BATCH * N_INP) + aswz(row, k, KB_INP)] = (bf16_t)x[i];
    }
    if (i < NWI) { const int h = i >> 6,  k = i & 63;   wib[bswz(h, k, KB_INP)] = (bf16_t)Wi[i]; }
    if (i < NWR) { const int h = i >> 11, k = i & 2047; wrb[bswz(h, k, KB_HID)] = (bf16_t)Wr[i]; }
    if (i < NS)  { sf0[i] = 1.0f; sb0[i] = (bf16_t)1.0f; }   // swizzle of all-ones is all-ones
    if (i < 32)  bar[i] = 0u;
  }
}

// per-wave loop-invariant context
struct WaveCtx {
  int    rsel;               // lanes>=16 hold rows v+8 in C/D layout
  int    gtid;               // global thread id (fp32 state slot base)
  float  Ah, Oh, Bh;         // per-column constants
  size_t whoff, wioff;       // B-frag bases (elements)
  size_t aoff0, aoff1;       // A-frag bases for the two M-tiles (hidden K)
  size_t xoff0, xoff1;       // A-frag bases for the two M-tiles (input K)
  size_t st0, st1;           // swizzled bf16 state-store bases for the two M-tiles
};

static __device__ __forceinline__ void rnn_step(
    int t, const WaveCtx& c,
    const bf16_t* __restrict__ whb, const bf16_t* __restrict__ wib,
    const bf16_t* __restrict__ xb,
    const bf16_t* __restrict__ Sb,  const float* __restrict__ Sf,
    bf16_t* __restrict__ Sbn,       float* __restrict__ Sfn) {
  // fused input projection: u = x_t @ Wi^T  (K = 64)
  v8f u0 = {}, u1 = {};
  const bf16_t* xt = xb + (size_t)t * (BATCH * N_INP);
#pragma unroll
  for (int kb = 0; kb < KB_INP; ++kb) {
    const size_t f = (size_t)kb * FRAG_ELEMS;
    v16bf b  = *(const v16bf*)(wib + c.wioff + f);
    v16bf a0 = *(const v16bf*)(xt  + c.xoff0 + f);
    v16bf a1 = *(const v16bf*)(xt  + c.xoff1 + f);
    u0 = WMMA_BF16(a0, b, u0);
    u1 = WMMA_BF16(a1, b, u1);
  }
  // recurrent GEMM: g = state @ Wh^T  (K = 2048, L2-resident, fragment-contiguous)
  v8f g0 = {}, g1 = {};
#pragma unroll 4
  for (int kb = 0; kb < KB_HID; ++kb) {
    const size_t f = (size_t)kb * FRAG_ELEMS;
    v16bf b  = *(const v16bf*)(whb + c.whoff + f);
    v16bf a0 = *(const v16bf*)(Sb  + c.aoff0 + f);
    v16bf a1 = *(const v16bf*)(Sb  + c.aoff1 + f);
    g0 = WMMA_BF16(a0, b, g0);
    g1 = WMMA_BF16(a1, b, g1);
  }
  // elementwise: new = A*cos(omega*s_old + u + bi) + g   (thread-private fp32 slots)
  const float* sp = Sf  + (size_t)c.gtid * 16;
  float*       dp = Sfn + (size_t)c.gtid * 16;
#pragma unroll
  for (int v = 0; v < 8; ++v) {
    const int r = v + c.rsel;                  // row within 16-row tile owned by this lane
    const float s0 = sp[v], s1 = sp[8 + v];
    const float n0 = c.Ah * __cosf(c.Oh * s0 + u0[v] + c.Bh) + g0[v];
    const float n1 = c.Ah * __cosf(c.Oh * s1 + u1[v] + c.Bh) + g1[v];
    dp[v] = n0;  dp[8 + v] = n1;
    Sbn[c.st0 + (size_t)r * 16] = (bf16_t)n0;  // swizzled bf16 state for next step's A
    Sbn[c.st1 + (size_t)r * 16] = (bf16_t)n1;
  }
}

// ---------- persistent scan kernel ----------
__global__ __launch_bounds__(256) void psrnn_scan(
    const bf16_t* __restrict__ whb, const bf16_t* __restrict__ wib,
    const bf16_t* __restrict__ wrb, const bf16_t* __restrict__ xb,
    const float* __restrict__ Avec, const float* __restrict__ Om,
    const float* __restrict__ Wib,  const float* __restrict__ Wrb,
    float* __restrict__ sf0, float* __restrict__ sf1,
    bf16_t* __restrict__ sb0, bf16_t* __restrict__ sb1,
    float* __restrict__ out, unsigned* __restrict__ bar) {
  const int lane = threadIdx.x & 31;
  const int wave = threadIdx.x >> 5;                    // 8 waves / block
  const int nt   = blockIdx.x * 4 + (wave & 3);         // global N-tile (0..127)
  const int mt0  = (wave >> 2) * 2;                     // M-tiles {mt0, mt0+1}
  const int col  = nt * 16 + (lane & 15);

  WaveCtx c;
  c.rsel = (lane >> 4) << 3;
  c.gtid = blockIdx.x * 256 + threadIdx.x;
  c.Ah = Avec[col];  c.Oh = Om[col];  c.Bh = Wib[col];
  c.whoff = (size_t)nt * KB_HID * FRAG_ELEMS + (size_t)lane * 16;
  c.wioff = (size_t)nt * KB_INP * FRAG_ELEMS + (size_t)lane * 16;
  c.aoff0 = (size_t)mt0       * KB_HID * FRAG_ELEMS + (size_t)lane * 16;
  c.aoff1 = (size_t)(mt0 + 1) * KB_HID * FRAG_ELEMS + (size_t)lane * 16;
  c.xoff0 = (size_t)mt0       * KB_INP * FRAG_ELEMS + (size_t)lane * 16;
  c.xoff1 = (size_t)(mt0 + 1) * KB_INP * FRAG_ELEMS + (size_t)lane * 16;
  {
    const int kbc = col >> 5, kk = col & 31;
    const int Ll  = ((kk >> 3) & 1) << 4;
    const int e   = (kk & 7) + ((kk >> 4) << 3);
    const size_t c0 = (size_t)kbc * FRAG_ELEMS + (size_t)Ll * 16 + e;
    c.st0 = (size_t)mt0       * KB_HID * FRAG_ELEMS + c0;
    c.st1 = (size_t)(mt0 + 1) * KB_HID * FRAG_ELEMS + c0;
  }

  unsigned* cnt = bar;
  unsigned* gen = bar + 16;

  // time loop unrolled by 2 so ping-pong buffers stay direct kernarg pointers
  for (int t = 0; t < T_STEPS; t += 2) {
    rnn_step(t,     c, whb, wib, xb, sb0, sf0, sb1, sf1);
    grid_barrier(cnt, gen);
    rnn_step(t + 1, c, whb, wib, xb, sb1, sf1, sb0, sf0);
    grid_barrier(cnt, gen);
  }

  // readout: out = state_final @ Wr^T + Wr_b  (final state in sb0; 16 tiles)
  const int gw = blockIdx.x * 8 + wave;
  if (gw < 16) {
    const int mt = gw >> 2, ot = gw & 3;
    const size_t ao = (size_t)mt * KB_HID * FRAG_ELEMS + (size_t)lane * 16;
    const size_t bo = (size_t)ot * KB_HID * FRAG_ELEMS + (size_t)lane * 16;
    v8f acc = {};
#pragma unroll 4
    for (int kb = 0; kb < KB_HID; ++kb) {
      const size_t f = (size_t)kb * FRAG_ELEMS;
      v16bf a = *(const v16bf*)(sb0 + ao + f);
      v16bf b = *(const v16bf*)(wrb + bo + f);
      acc = WMMA_BF16(a, b, acc);
    }
    const int oc = ot * 16 + (lane & 15);
    const float bb = Wrb[oc];
#pragma unroll
    for (int v = 0; v < 8; ++v) {
      const int orow = mt * 16 + v + c.rsel;
      out[(size_t)orow * N_OUT + oc] = acc[v] + bb;
    }
  }
}

extern "C" void kernel_launch(void* const* d_in, const int* in_sizes, int n_in,
                              void* d_out, int out_size, void* d_ws, size_t ws_size,
                              hipStream_t stream) {
  (void)in_sizes; (void)n_in; (void)out_size; (void)ws_size;
  const float* x    = (const float*)d_in[0];
  const float* Wi_w = (const float*)d_in[1];
  const float* Wi_b = (const float*)d_in[2];
  const float* Wh_w = (const float*)d_in[3];
  const float* A    = (const float*)d_in[4];
  const float* om   = (const float*)d_in[5];
  const float* Wr_w = (const float*)d_in[6];
  const float* Wr_b = (const float*)d_in[7];
  float* out = (float*)d_out;

  char* ws = (char*)d_ws;
  size_t off = 0;
  auto alloc = [&](size_t bytes) -> void* {
    void* p = ws + off;
    off = (off + bytes + 255) & ~(size_t)255;
    return p;
  };
  bf16_t*   whb = (bf16_t*)alloc((size_t)N_HID * N_HID * 2);
  bf16_t*   wib = (bf16_t*)alloc((size_t)N_HID * N_INP * 2);
  bf16_t*   wrb = (bf16_t*)alloc((size_t)N_OUT * N_HID * 2);
  bf16_t*   xbb = (bf16_t*)alloc((size_t)T_STEPS * BATCH * N_INP * 2);
  float*    sf0 = (float*)alloc((size_t)BATCH * N_HID * 4);
  float*    sf1 = (float*)alloc((size_t)BATCH * N_HID * 4);
  bf16_t*   sb0 = (bf16_t*)alloc((size_t)BATCH * N_HID * 2);
  bf16_t*   sb1 = (bf16_t*)alloc((size_t)BATCH * N_HID * 2);
  unsigned* bar = (unsigned*)alloc(32 * sizeof(unsigned));

  psrnn_prep<<<4096, 1024, 0, stream>>>(Wh_w, Wi_w, Wr_w, x,
                                        whb, wib, wrb, xbb, sf0, sb0, bar);
  psrnn_scan<<<NBLK, 256, 0, stream>>>(whb, wib, wrb, xbb,
                                       A, om, Wi_b, Wr_b,
                                       sf0, sf1, sb0, sb1, out, bar);
}
